// MambaLayer_60292750901854
// MI455X (gfx1250) — matrix operand order repaired and checked by reading, not compiled, not run on hardware
//
#include <hip/hip_runtime.h>
#include <hip/hip_bf16.h>

// ---- problem constants ----
#define B_  2
#define C_  384
#define L_  8192           // 8*32*32
#define DI  768            // d_inner
#define DS  16             // d_state
#define DTR 24             // dt_rank
#define DCV 4              // d_conv
#define M_  (B_*L_)        // 16384 rows
#define NCH 64             // scan chunks
#define LC  (L_/NCH)       // 128 per chunk

typedef _Float16 h8     __attribute__((ext_vector_type(8)));
typedef _Float16 v16h   __attribute__((ext_vector_type(16)));
typedef float    v8f    __attribute__((ext_vector_type(8)));
typedef unsigned int  u32x4 __attribute__((ext_vector_type(4)));
typedef int           i32x4 __attribute__((ext_vector_type(4)));
typedef int           i32x8 __attribute__((ext_vector_type(8)));

// TDM builtin arity differs between toolchains; the TDM header ships only with
// the 6-arg (therock) toolchain, so use it as a toolchain fingerprint.
#if __has_include(<hip/amd_detail/amd_gfx1250_TDM.h>)
#define TDM_6ARG 1
#else
#define TDM_6ARG 0
#endif

static __device__ __forceinline__ void tdm_issue(u32x4 g0, i32x8 g1) {
#if TDM_6ARG
  __builtin_amdgcn_tensor_load_to_lds(g0, g1, (i32x4)0, (i32x4)0, (i32x8)0, 0);
#else
  __builtin_amdgcn_tensor_load_to_lds(g0, g1, (i32x4)0, (i32x4)0, 0);
#endif
}

// 2-D TDM tile load: 64 rows x 32 f16, row stride K elements, into LDS at
// ldsOff with 16B pad after each 64B row (LDS row stride = 80B).
static __device__ __forceinline__ void tdm_tile_load(const _Float16* g, unsigned ldsOff, int K) {
  unsigned long long ga = (unsigned long long)(uintptr_t)g;
  u32x4 g0;
  g0.x = 1u;                                   // count=1 valid descriptor
  g0.y = ldsOff;                               // LDS byte address
  g0.z = (unsigned)ga;                         // global addr [31:0]
  g0.w = (unsigned)((ga >> 32) & 0x01FFFFFFu) | 0x80000000u;  // addr[56:32] | type=2
  i32x8 g1;
  g1[0] = 0x06D10000;          // data_size=2B, pad_enable, interval=16dw, amount=4dw
  g1[1] = (int)(32u << 16);    // tensor_dim0 = 32 (lo16 at bits 63:48)
  g1[2] = (int)(64u << 16);    // tensor_dim0 hi=0 | tensor_dim1 = 64
  g1[3] = (int)(32u << 16);    // tensor_dim1 hi=0 | tile_dim0 = 32
  g1[4] = 64;                  // tile_dim1 = 64, tile_dim2 = 0
  g1[5] = K;                   // tensor_dim0_stride (elements)
  g1[6] = 0;
  g1[7] = 0;
  tdm_issue(g0, g1);
}

static __device__ __forceinline__ unsigned lds_off(const void* p) {
  return (unsigned)(unsigned long long)(uintptr_t)p;   // low 32 bits = LDS offset
}

static __device__ __forceinline__ v8f wmma32(v16h a, v16h b, v8f c) {
  return __builtin_amdgcn_wmma_f32_16x16x32_f16(false, a, false, b, (short)0, c,
                                                false, false);
}
static __device__ __forceinline__ float silu_f(float v) {
  return v / (1.0f + __expf(-v));
}

// ---------- utility: f32 -> f16 convert / zero fill ----------
__global__ void cvt_f16_kernel(const float* __restrict__ s, _Float16* __restrict__ d, int n) {
  int i = blockIdx.x * 256 + threadIdx.x;
  if (i < n) d[i] = (_Float16)s[i];
}
__global__ void fill0_f16_kernel(_Float16* __restrict__ d, int n) {
  int i = blockIdx.x * 256 + threadIdx.x;
  if (i < n) d[i] = (_Float16)0.0f;
}

// ---------- LayerNorm over C; x is (B,C,L) channel-major -> xn_h (B*L, C) f16 ----------
__global__ void ln_kernel(const float* __restrict__ x, const float* __restrict__ lw,
                          const float* __restrict__ lb, _Float16* __restrict__ xn) {
  __shared__ float redS[8*32], redS2[8*32], meanL[32], rstdL[32];
  __shared__ _Float16 tile[32 * 392];
  const int b  = blockIdx.x / (L_/32);
  const int l0 = (blockIdx.x % (L_/32)) * 32;
  const int w  = threadIdx.x >> 5, lane = threadIdx.x & 31;
  const float* xb = x + (size_t)b * C_ * L_ + l0 + lane;
  float s = 0.f, s2 = 0.f;
  for (int c = w; c < C_; c += 8) { float v = xb[(size_t)c * L_]; s += v; s2 += v*v; }
  redS [w*32+lane] = s;  redS2[w*32+lane] = s2;
  __syncthreads();
  if (w == 0) {
    float a = 0.f, a2 = 0.f;
    #pragma unroll
    for (int j = 0; j < 8; ++j) { a += redS[j*32+lane]; a2 += redS2[j*32+lane]; }
    float mu  = a * (1.0f / C_);
    float var = a2 * (1.0f / C_) - mu * mu;
    meanL[lane] = mu;  rstdL[lane] = rsqrtf(var + 1e-5f);
  }
  __syncthreads();
  const float mu = meanL[lane], rs = rstdL[lane];
  for (int c = w; c < C_; c += 8) {
    float v = xb[(size_t)c * L_];
    tile[lane*392 + c] = (_Float16)((v - mu) * rs * lw[c] + lb[c]);
  }
  __syncthreads();
  for (int idx = threadIdx.x; idx < 32*C_; idx += 256) {
    int lr = idx / C_, c = idx % C_;
    xn[(size_t)(b*L_ + l0 + lr) * C_ + c] = tile[lr*392 + c];
  }
}

// ---------- WMMA GEMM with TDM-staged, double-buffered LDS tiles ----------
// out[m,n] = sum_k A[m,k]*Bw[n,k];   64x64 tile per 256-thread block.
// LDS tile: 64 rows x 32 f16, row stride 40 halves (80B; TDM pads 16B/row).
#define TROW 40
#define TBYTES (64 * TROW * 2)        // 5120B per tile buffer
template<bool OUT_F16>
__global__ void gemm_wmma_kernel(const _Float16* __restrict__ A,
                                 const _Float16* __restrict__ Bw,
                                 void* __restrict__ Cout, int M, int N, int K) {
  __shared__ _Float16 sA[2][64 * TROW];
  __shared__ _Float16 sB[2][64 * TROW];
  const int wave = threadIdx.x >> 5, lane = threadIdx.x & 31;
  const int half = lane >> 4,       lm   = lane & 15;
  const int mtile = blockIdx.x * 64, ntile = blockIdx.y * 64;
  const int mstrip = (wave & 3) * 16;       // this wave's 16-row strip
  const int njloc  = (wave >> 2) * 32;      // this wave's two 16-col tiles
  const unsigned aOff = lds_off(&sA[0][0]);
  const unsigned bOff = lds_off(&sB[0][0]);
  const int nk = K / 32;

  v8f acc0 = {0,0,0,0,0,0,0,0};
  v8f acc1 = {0,0,0,0,0,0,0,0};

  auto issue = [&](int i, int buf) {
    tdm_tile_load(A  + (size_t)mtile * K + i * 32, aOff + (unsigned)buf * TBYTES, K);
    tdm_tile_load(Bw + (size_t)ntile * K + i * 32, bOff + (unsigned)buf * TBYTES, K);
  };

  if (threadIdx.x < 32) issue(0, 0);        // wave 0 drives the TDM
  for (int i = 0; i < nk; ++i) {
    const int cur = i & 1;
    if (threadIdx.x < 32) {
      if (i + 1 < nk) {
        issue(i + 1, cur ^ 1);              // prefetch next K-step into other buffer
        __builtin_amdgcn_s_wait_tensorcnt(2);   // current pair landed
      } else {
        __builtin_amdgcn_s_wait_tensorcnt(0);
      }
    }
    __syncthreads();                        // publish LDS tile to all waves
    // A fragment: 16 contiguous K per lane (half-wave K split), two 8-half chunks
    const _Float16* sa = &sA[cur][(mstrip + lm) * TROW];
    h8 lo = *(const h8*)(sa + half * 16);
    h8 hi = *(const h8*)(sa + half * 16 + 8);
    v16h af;
    #pragma unroll
    for (int j = 0; j < 8; ++j) { af[j] = lo[j]; af[j+8] = hi[j]; }
    // B fragments: lane<->N, 16 contiguous K per lane (half-wave K split)
    v16h b0 = *(const v16h*)(&sB[cur][(njloc      + lm) * TROW + half * 16]);
    v16h b1 = *(const v16h*)(&sB[cur][(njloc + 16 + lm) * TROW + half * 16]);
    acc0 = wmma32(af, b0, acc0);
    acc1 = wmma32(af, b1, acc1);
    __syncthreads();                        // buffer reusable for next prefetch
  }

  const int n0 = ntile + njloc + lm, n1 = n0 + 16;
  #pragma unroll
  for (int i = 0; i < 8; ++i) {
    int r = mtile + mstrip + half * 8 + i;
    if (OUT_F16) {
      _Float16* o = (_Float16*)Cout;
      o[(size_t)r * N + n0] = (_Float16)acc0[i];
      o[(size_t)r * N + n1] = (_Float16)acc1[i];
    } else {
      float* o = (float*)Cout;
      o[(size_t)r * N + n0] = acc0[i];
      o[(size_t)r * N + n1] = acc1[i];
    }
  }
}

// ---------- depthwise causal conv(4) + SiLU: xz_h[:, :DI] -> xc_h ----------
__global__ void conv_kernel(const _Float16* __restrict__ xz, const float* __restrict__ cw,
                            const float* __restrict__ cb, _Float16* __restrict__ xc) {
  int idx = blockIdx.x * 256 + threadIdx.x;     // m*DI + d
  if (idx >= M_ * DI) return;
  int d = idx % DI, m = idx / DI;
  int b = m / L_, l = m % L_;
  float acc = cb[d];
  #pragma unroll
  for (int j = 0; j < DCV; ++j) {
    int ls = l + j - (DCV - 1);
    if (ls >= 0)
      acc += cw[d*DCV + j] * (float)xz[(size_t)(b*L_ + ls) * (2*DI) + d];
  }
  xc[idx] = (_Float16)silu_f(acc);
}

// ---------- dt = softplus(x_dbl[:, :24] @ W_dt^T + b_dt) ----------
__global__ void dt_kernel(const float* __restrict__ xdbl, const float* __restrict__ Wdt,
                          const float* __restrict__ bdt, _Float16* __restrict__ dt) {
  int idx = blockIdx.x * 256 + threadIdx.x;     // m*DI + d
  if (idx >= M_ * DI) return;
  int d = idx % DI, m = idx / DI;
  const float* xr = xdbl + (size_t)m * 64;
  const float* wr = Wdt + d * DTR;
  float acc = bdt[d];
  #pragma unroll
  for (int r = 0; r < DTR; ++r) acc += xr[r] * wr[r];
  float sp = (acc > 20.0f) ? acc : log1pf(__expf(acc));
  dt[idx] = (_Float16)sp;
}

// ---------- scan phase 1: per (b,d,chunk) local scan from h=0 ----------
__global__ void scan1_kernel(const _Float16* __restrict__ dt, const _Float16* __restrict__ xc,
                             const float* __restrict__ xdbl, const float* __restrict__ Alog,
                             float* __restrict__ Sbuf, float* __restrict__ qbuf) {
  int t = blockIdx.x * 256 + threadIdx.x;       // ((b*NCH+ch)*DI + d)
  if (t >= B_ * NCH * DI) return;
  int d = t % DI, ch = (t / DI) % NCH, b = t / (DI * NCH);
  float A[DS], q[DS];
  #pragma unroll
  for (int n = 0; n < DS; ++n) { A[n] = -__expf(Alog[d*DS + n]); q[n] = 0.f; }
  float S = 0.f;
  int row0 = b * L_ + ch * LC;
  for (int i = 0; i < LC; ++i) {
    int row = row0 + i;
    float dtv = (float)dt[(size_t)row * DI + d];
    float xv  = (float)xc[(size_t)row * DI + d];
    float dx  = dtv * xv;
    const float* xd = xdbl + (size_t)row * 64;
    #pragma unroll
    for (int n = 0; n < DS; ++n)
      q[n] = __expf(dtv * A[n]) * q[n] + dx * xd[DTR + n];
    S += dtv;
  }
  Sbuf[t] = S;
  float* qp = qbuf + (size_t)t * DS;
  #pragma unroll
  for (int n = 0; n < DS; ++n) qp[n] = q[n];
}

// ---------- scan phase 2: combine chunk carries (serial over NCH) ----------
__global__ void scan2_kernel(const float* __restrict__ Sbuf, const float* __restrict__ qbuf,
                             const float* __restrict__ Alog, float* __restrict__ hin) {
  int t = blockIdx.x * 256 + threadIdx.x;       // (b*DI+d)*DS + n
  if (t >= B_ * DI * DS) return;
  int n = t % DS, d = (t / DS) % DI, b = t / (DS * DI);
  float A = -__expf(Alog[d*DS + n]);
  float h = 0.f;
  for (int c = 0; c < NCH; ++c) {
    size_t idx = (size_t)((b*NCH + c) * DI + d);
    hin[idx * DS + n] = h;
    h = __expf(A * Sbuf[idx]) * h + qbuf[idx * DS + n];
  }
}

// ---------- scan phase 3: replay with correct h_in; fuse +x*D and *silu(z) ----------
__global__ void scan3_kernel(const _Float16* __restrict__ dt, const _Float16* __restrict__ xc,
                             const float* __restrict__ xdbl, const _Float16* __restrict__ xz,
                             const float* __restrict__ Alog, const float* __restrict__ Dp,
                             const float* __restrict__ hin, _Float16* __restrict__ yh) {
  int t = blockIdx.x * 256 + threadIdx.x;
  if (t >= B_ * NCH * DI) return;
  int d = t % DI, ch = (t / DI) % NCH, b = t / (DI * NCH);
  float A[DS], h[DS];
  const float* hp = hin + (size_t)t * DS;
  #pragma unroll
  for (int n = 0; n < DS; ++n) { A[n] = -__expf(Alog[d*DS + n]); h[n] = hp[n]; }
  const float Dd = Dp[d];
  int row0 = b * L_ + ch * LC;
  for (int i = 0; i < LC; ++i) {
    int row = row0 + i;
    float dtv = (float)dt[(size_t)row * DI + d];
    float xv  = (float)xc[(size_t)row * DI + d];
    float dx  = dtv * xv;
    const float* xd = xdbl + (size_t)row * 64;
    float y = 0.f;
    #pragma unroll
    for (int n = 0; n < DS; ++n) {
      h[n] = __expf(dtv * A[n]) * h[n] + dx * xd[DTR + n];
      y += h[n] * xd[DTR + DS + n];
    }
    y += xv * Dd;
    float z = (float)xz[(size_t)row * (2*DI) + DI + d];
    yh[(size_t)row * DI + d] = (_Float16)(y * silu_f(z));
  }
}

// ---------- final transpose (B*L, C) f32 -> (B, C, L) f32 ----------
__global__ void transpose_kernel(const float* __restrict__ src, float* __restrict__ dst) {
  __shared__ float tile[32][33];
  int b  = blockIdx.x;
  int l0 = blockIdx.y * 32;
  int c0 = blockIdx.z * 32;
  int tx = threadIdx.x, ty = threadIdx.y;
  #pragma unroll
  for (int i = 0; i < 4; ++i)
    tile[ty + i*8][tx] = src[(size_t)(b*L_ + l0 + ty + i*8) * C_ + c0 + tx];
  __syncthreads();
  #pragma unroll
  for (int i = 0; i < 4; ++i)
    dst[(size_t)b * C_ * L_ + (size_t)(c0 + ty + i*8) * L_ + l0 + tx] = tile[tx][ty + i*8];
}

extern "C" void kernel_launch(void* const* d_in, const int* in_sizes, int n_in,
                              void* d_out, int out_size, void* d_ws, size_t ws_size,
                              hipStream_t stream) {
  const float* x      = (const float*)d_in[0];
  const float* ln_w   = (const float*)d_in[1];
  const float* ln_b   = (const float*)d_in[2];
  const float* W_in   = (const float*)d_in[3];
  const float* conv_w = (const float*)d_in[4];
  const float* conv_b = (const float*)d_in[5];
  const float* W_xprj = (const float*)d_in[6];
  const float* W_dt   = (const float*)d_in[7];
  const float* b_dt   = (const float*)d_in[8];
  const float* A_log  = (const float*)d_in[9];
  const float* Dp     = (const float*)d_in[10];
  const float* W_out  = (const float*)d_in[11];
  float* out = (float*)d_out;

  // workspace layout (256B aligned); xn region reused for x_dbl, xz region for out_tmp
  char*  ws = (char*)d_ws;
  size_t o  = 0;
  auto alloc = [&](size_t bytes) { size_t r = o; o = (o + bytes + 255) & ~(size_t)255; return r; };
  size_t oWin  = alloc((size_t)2*DI*C_*2);        // W_in f16   (1536x384)
  size_t oWxp  = alloc((size_t)64*DI*2);          // W_xproj f16 padded to 64 rows
  size_t oWout = alloc((size_t)C_*DI*2);          // W_out f16  (384x768)
  size_t oXn   = alloc((size_t)M_*C_*2);          // xn f16; later x_dbl f32 (M x 64) fits
  size_t oXz   = alloc((size_t)M_*2*DI*2);        // xz f16; later out_tmp f32 (M x 384) fits
  size_t oXc   = alloc((size_t)M_*DI*2);          // xc f16
  size_t oDt   = alloc((size_t)M_*DI*2);          // dt f16
  size_t oS    = alloc((size_t)B_*NCH*DI*4);      // chunk dt-sums
  size_t oQ    = alloc((size_t)B_*NCH*DI*DS*4);   // chunk local states
  size_t oHin  = alloc((size_t)B_*NCH*DI*DS*4);   // chunk entry states
  size_t oY    = alloc((size_t)M_*DI*2);          // gated y f16
  (void)ws_size;

  _Float16* W_in_h  = (_Float16*)(ws + oWin);
  _Float16* W_xp_h  = (_Float16*)(ws + oWxp);
  _Float16* W_out_h = (_Float16*)(ws + oWout);
  _Float16* xn_h    = (_Float16*)(ws + oXn);
  float*    x_dbl   = (float*)   (ws + oXn);      // reuse after GEMM1
  _Float16* xz_h    = (_Float16*)(ws + oXz);
  float*    out_tmp = (float*)   (ws + oXz);      // reuse after scan3
  _Float16* xc_h    = (_Float16*)(ws + oXc);
  _Float16* dt_h    = (_Float16*)(ws + oDt);
  float*    Sbuf    = (float*)   (ws + oS);
  float*    qbuf    = (float*)   (ws + oQ);
  float*    hin     = (float*)   (ws + oHin);
  _Float16* y_h     = (_Float16*)(ws + oY);

  auto blks = [](int n) { return (n + 255) / 256; };

  // weights -> fp16 (W_xproj padded with zero rows 56..63 for WMMA B-tile reads)
  cvt_f16_kernel<<<blks(2*DI*C_), 256, 0, stream>>>(W_in, W_in_h, 2*DI*C_);
  cvt_f16_kernel<<<blks(56*DI), 256, 0, stream>>>(W_xprj, W_xp_h, 56*DI);
  fill0_f16_kernel<<<blks(8*DI), 256, 0, stream>>>(W_xp_h + 56*DI, 8*DI);
  cvt_f16_kernel<<<blks(C_*DI), 256, 0, stream>>>(W_out, W_out_h, C_*DI);

  // LayerNorm (LDS-tiled: x is channel-major)
  ln_kernel<<<B_ * (L_/32), 256, 0, stream>>>(x, ln_w, ln_b, xn_h);

  // GEMM1: xz = xn @ W_in^T   (16384 x 384 x 1536)
  gemm_wmma_kernel<true><<<dim3(M_/64, (2*DI)/64), 256, 0, stream>>>(
      xn_h, W_in_h, (void*)xz_h, M_, 2*DI, C_);

  // depthwise conv + SiLU
  conv_kernel<<<blks(M_*DI), 256, 0, stream>>>(xz_h, conv_w, conv_b, xc_h);

  // GEMM2: x_dbl = xc @ W_xproj^T  (16384 x 768 x 64-padded), f32 out
  gemm_wmma_kernel<false><<<dim3(M_/64, 1), 256, 0, stream>>>(
      xc_h, W_xp_h, (void*)x_dbl, M_, 64, DI);

  // dt projection + softplus (K=24 scalar)
  dt_kernel<<<blks(M_*DI), 256, 0, stream>>>(x_dbl, W_dt, b_dt, dt_h);

  // chunked parallel selective scan
  scan1_kernel<<<blks(B_*NCH*DI), 256, 0, stream>>>(dt_h, xc_h, x_dbl, A_log, Sbuf, qbuf);
  scan2_kernel<<<blks(B_*DI*DS), 256, 0, stream>>>(Sbuf, qbuf, A_log, hin);
  scan3_kernel<<<blks(B_*NCH*DI), 256, 0, stream>>>(dt_h, xc_h, x_dbl, xz_h, A_log, Dp,
                                                    hin, y_h);

  // GEMM3: out_tmp = y @ W_out^T  (16384 x 768 x 384), f32 out
  gemm_wmma_kernel<false><<<dim3(M_/64, C_/64), 256, 0, stream>>>(
      y_h, W_out_h, (void*)out_tmp, M_, C_, DI);

  // (B*L, C) -> (B, C, L)
  transpose_kernel<<<dim3(B_, L_/32, C_/32), dim3(32, 8), 0, stream>>>(out_tmp, out);
  (void)in_sizes; (void)n_in; (void)out_size;
}
